// WindowAttention_66305705116082
// MI455X (gfx1250) — matrix operand order repaired and checked by reading, not compiled, not run on hardware
//
#include <hip/hip_runtime.h>
#include <hip/hip_bf16.h>
#include <math.h>

// Swin shifted-window attention, fully fused, one 8x8 window per 256-thread block.
// All GEMMs via v_wmma_f32_16x16x32_f16 (wave32 CDNA5 WMMA).
// - weights pre-converted to f16 + pre-swizzled into B-fragment order (d_ws)
// - all LDS row strides padded (+4 dwords) so fragment loads are bank-conflict-free
// - softmax probabilities P overlay the score buffer S in place -> 152 KB LDS,
//   two blocks per 320 KB WGP.

typedef __attribute__((ext_vector_type(16))) _Float16 v16h;
typedef __attribute__((ext_vector_type(8)))  float    v8f;

#define D_MODEL 128
#define QKV_N   384
#define HD      32
#define SCALE   0.17677669529663687f   // 1/sqrt(32)

#define WQKV_ELEMS (D_MODEL * QKV_N)   // 49152
#define WOUT_ELEMS (D_MODEL * D_MODEL) // 16384

// padded LDS strides (halfwords / floats)
#define XS_STR   136    // 128 + 8 halfwords
#define QKV_STR  392    // 384 + 8 halfwords
#define S_STR    68     // 64 + 4 floats
#define P_STR    136    // S row reinterpreted as halfwords (68 floats = 136 halfs)
#define O_STR    136

// ---- fragment loaders (wave32 WMMA 16x16x32 f16 layouts) -------------------
__device__ inline v16h lds_afrag(const _Float16* m, int stride, int m0, int k0) {
    const int lane = threadIdx.x & 31;
    const _Float16* p = m + (m0 + (lane & 15)) * stride + k0 + 16 * (lane >> 4);
    v16h a;
#pragma unroll
    for (int e = 0; e < 8; ++e) a[e] = p[e];
#pragma unroll
    for (int e = 0; e < 8; ++e) a[8 + e] = p[8 + e];
    return a;
}

// B fragment where source rows are the N dimension (i.e. source is transposed)
__device__ inline v16h lds_bfragT(const _Float16* m, int stride, int n0, int k0) {
    const int lane = threadIdx.x & 31;
    const _Float16* p = m + (n0 + (lane & 15)) * stride + k0 + 16 * (lane >> 4);
    v16h b;
#pragma unroll
    for (int e = 0; e < 16; ++e) b[e] = p[e];
    return b;
}

// B fragment from row-major [K][N] f16 LDS (per-element strided)
__device__ inline v16h lds_bfrag(const _Float16* m, int stride, int k0, int n0) {
    const int lane = threadIdx.x & 31;
    const _Float16* p = m + (k0 + 16 * (lane >> 4)) * stride + n0 + (lane & 15);
    v16h b;
#pragma unroll
    for (int e = 0; e < 16; ++e) b[e] = p[e * stride];
    return b;
}

// B fragment from pre-swizzled global weights [kTile][n][kk] -> 2x b128 loads
__device__ inline v16h glb_bfrag_swz(const _Float16* w, int nTotal, int kTile, int n0) {
    const int lane = threadIdx.x & 31;
    const _Float16* p = w + ((size_t)(kTile * nTotal + n0 + (lane & 15)) << 5) + 16 * (lane >> 4);
    v16h b;
#pragma unroll
    for (int e = 0; e < 16; ++e) b[e] = p[e];
    return b;
}

__device__ inline v8f wmma32(v16h a, v16h b, v8f c) {
    return __builtin_amdgcn_wmma_f32_16x16x32_f16(
        false, a, false, b, (short)0, c, false, false);
}

// ---- one-time weight convert + swizzle: f32 [K][N] -> f16 [K/32][N][32] ----
__global__ __launch_bounds__(256)
void weight_swizzle_kernel(const float* __restrict__ w_qkv,
                           const float* __restrict__ w_out,
                           _Float16* __restrict__ wqkv_h,
                           _Float16* __restrict__ wout_h) {
    int id = blockIdx.x * 256 + threadIdx.x;     // 0 .. 65535
    if (id < WQKV_ELEMS) {
        int k = id / QKV_N, n = id % QKV_N;
        wqkv_h[(((k >> 5) * QKV_N + n) << 5) + (k & 31)] = (_Float16)w_qkv[id];
    } else {
        int id2 = id - WQKV_ELEMS;               // < 16384
        int k = id2 / D_MODEL, n = id2 % D_MODEL;
        wout_h[(((k >> 5) * D_MODEL + n) << 5) + (k & 31)] = (_Float16)w_out[id2];
    }
}

// ---- LDS layout (bytes) ----------------------------------------------------
//   [0      ,  50176)  QKV f16 [64][392]
//   [50176  , 119808)  S   f32 [4][64][68]   (P f16 overlays in place)
//   [119808 , 137216)  Xs  f16 [64][136]
//   [137216 , 154624)  O   f16 [64][136]
//   [154624 , 155648)  pos f32 [225]
#define OFF_QKV 0
#define OFF_S   50176
#define OFF_XS  119808
#define OFF_O   137216
#define OFF_POS 154624
#define SMEM_BYTES 155648   // 152 KB -> 2 blocks / 320 KB WGP

__global__ __launch_bounds__(256)
void swin_window_attn_kernel(const float* __restrict__ x,
                             const _Float16* __restrict__ wqkv_h,
                             const float* __restrict__ pos,
                             const _Float16* __restrict__ wout_h,
                             const float* __restrict__ b_out,
                             float* __restrict__ out) {
    extern __shared__ char smem[];
    _Float16* QKV  = (_Float16*)(smem + OFF_QKV);
    float*    S    = (float*)   (smem + OFF_S);
    _Float16* P    = (_Float16*)(smem + OFF_S);     // overlays S in place
    _Float16* Xs   = (_Float16*)(smem + OFF_XS);
    _Float16* O    = (_Float16*)(smem + OFF_O);
    float*    posl = (float*)   (smem + OFF_POS);

    const int tid  = threadIdx.x;
    const int wave = tid >> 5;
    const int lane = tid & 31;
    const int l16  = lane & 15;
    const int lh   = lane >> 4;

    const int b   = blockIdx.x >> 10;        // batch
    const int win = blockIdx.x & 1023;
    const int wy  = win >> 5;                // window row  (0..31)
    const int wx  = win & 31;                // window col  (0..31)

    if (tid < 225) posl[tid] = pos[tid];

    // ---- load x window with cyclic shift (-4,-4) folded into the index ----
    const float* xb = x + (size_t)b * 256 * 256 * 128;
#pragma unroll
    for (int i = 0; i < 32; ++i) {
        int idx = tid + i * 256;             // 0..8191 = token*128 + d
        int t = idx >> 7, d = idx & 127;
        int gr = ((wy << 3) + (t >> 3) + 4) & 255;
        int gc = ((wx << 3) + (t & 7) + 4) & 255;
        Xs[t * XS_STR + d] = (_Float16)xb[((size_t)((gr << 8) + gc) << 7) + d];
    }
    __syncthreads();

    // ---- stage 1: QKV = Xs(64x128) @ Wqkv(128x384) ------------------------
    for (int nti = 0; nti < 3; ++nti) {
        const int n0 = (wave * 3 + nti) * 16;
        v8f acc0 = 0, acc1 = 0, acc2 = 0, acc3 = 0;
#pragma unroll
        for (int kc = 0; kc < 4; ++kc) {
            v16h bf = glb_bfrag_swz(wqkv_h, QKV_N, kc, n0);
            acc0 = wmma32(lds_afrag(Xs, XS_STR,  0, kc * 32), bf, acc0);
            acc1 = wmma32(lds_afrag(Xs, XS_STR, 16, kc * 32), bf, acc1);
            acc2 = wmma32(lds_afrag(Xs, XS_STR, 32, kc * 32), bf, acc2);
            acc3 = wmma32(lds_afrag(Xs, XS_STR, 48, kc * 32), bf, acc3);
        }
        const v8f* accs[4] = {&acc0, &acc1, &acc2, &acc3};
#pragma unroll
        for (int mt = 0; mt < 4; ++mt)
#pragma unroll
            for (int r = 0; r < 8; ++r)
                QKV[(mt * 16 + r + 8 * lh) * QKV_STR + n0 + l16] = (_Float16)(*accs[mt])[r];
    }
    __syncthreads();

    // ---- stage 2: S = Q . K^T per head (head_dim 32 == one WMMA K-step) ---
    {
        const int h = wave >> 1;
        const int msub = wave & 1;
        float* Sh = S + h * 64 * S_STR;
#pragma unroll
        for (int mi = 0; mi < 2; ++mi) {
            const int m0 = (msub * 2 + mi) * 16;
            v16h af = lds_afrag(QKV, QKV_STR, m0, h * HD);               // Q rows
#pragma unroll
            for (int nt = 0; nt < 4; ++nt) {
                v16h bf = lds_bfragT(QKV, QKV_STR, nt * 16, 128 + h * HD); // K rows
                v8f acc = 0;
                acc = wmma32(af, bf, acc);
#pragma unroll
                for (int r = 0; r < 8; ++r)
                    Sh[(m0 + r + 8 * lh) * S_STR + nt * 16 + l16] = acc[r];
            }
        }
    }
    __syncthreads();

    // ---- stage 3: softmax with rel-pos bias + shift masks (in place) ------
    {
        const int h = tid >> 6, i = tid & 63;
        const int ri = i >> 3, ci = i & 7;
        const bool mUL = (wy == 31), mLR = (wx == 31);
        float*    Sh = S + (h * 64 + i) * S_STR;   // thread-private row
        _Float16* Ph = P + (h * 64 + i) * P_STR;   // same bytes, f16 view
        float mx = -1e30f;
        for (int j = 0; j < 64; ++j) {
            int rj = j >> 3, cj = j & 7;
            float v = Sh[j] * SCALE + posl[(rj - ri + 7) * 15 + (cj - ci + 7)];
            if (mUL && ((ri >= 4) != (rj >= 4))) v = -1e30f;
            if (mLR && ((ci >= 4) != (cj >= 4))) v = -1e30f;
            Sh[j] = v;
            mx = fmaxf(mx, v);
        }
        float sum = 0.f;
        for (int j = 0; j < 64; ++j) {
            float e = __expf(Sh[j] - mx);   // read f32 elem j, then clobber
            sum += e;                        // bytes [2j,2j+2) = float j/2 <= j
            Ph[j] = (_Float16)e;
        }
        float inv = 1.0f / sum;
        for (int j = 0; j < 64; ++j)
            Ph[j] = (_Float16)((float)Ph[j] * inv);
    }
    __syncthreads();

    // ---- stage 4: O = P(64x64) @ V(64x32) per head ------------------------
    {
        const int h = wave >> 1;
        const int n0c = (wave & 1) * 16;     // 0 or 16 within head_dim
        const _Float16* Ph = P + h * 64 * P_STR;
        v8f acc0 = 0, acc1 = 0, acc2 = 0, acc3 = 0;
#pragma unroll
        for (int kc = 0; kc < 2; ++kc) {
            v16h bf = lds_bfrag(QKV, QKV_STR, kc * 32, 256 + h * HD + n0c); // V
            acc0 = wmma32(lds_afrag(Ph, P_STR,  0, kc * 32), bf, acc0);
            acc1 = wmma32(lds_afrag(Ph, P_STR, 16, kc * 32), bf, acc1);
            acc2 = wmma32(lds_afrag(Ph, P_STR, 32, kc * 32), bf, acc2);
            acc3 = wmma32(lds_afrag(Ph, P_STR, 48, kc * 32), bf, acc3);
        }
        const v8f* accs[4] = {&acc0, &acc1, &acc2, &acc3};
#pragma unroll
        for (int mt = 0; mt < 4; ++mt)
#pragma unroll
            for (int r = 0; r < 8; ++r)
                O[(mt * 16 + r + 8 * lh) * O_STR + h * HD + n0c + l16] =
                    (_Float16)(*accs[mt])[r];
    }
    __syncthreads();

    // ---- stage 5: out = O(64x128) @ Wout(128x128) + b, store with +4 roll -
    {
        const int n0 = wave * 16;
        v8f acc0 = 0, acc1 = 0, acc2 = 0, acc3 = 0;
#pragma unroll
        for (int kc = 0; kc < 4; ++kc) {
            v16h bf = glb_bfrag_swz(wout_h, D_MODEL, kc, n0);
            acc0 = wmma32(lds_afrag(O, O_STR,  0, kc * 32), bf, acc0);
            acc1 = wmma32(lds_afrag(O, O_STR, 16, kc * 32), bf, acc1);
            acc2 = wmma32(lds_afrag(O, O_STR, 32, kc * 32), bf, acc2);
            acc3 = wmma32(lds_afrag(O, O_STR, 48, kc * 32), bf, acc3);
        }
        const int n = n0 + l16;
        const float bias = b_out[n];
        float* ob = out + (size_t)b * 256 * 256 * 128;
        const v8f* accs[4] = {&acc0, &acc1, &acc2, &acc3};
#pragma unroll
        for (int mt = 0; mt < 4; ++mt)
#pragma unroll
            for (int r = 0; r < 8; ++r) {
                int t = mt * 16 + r + 8 * lh;
                int gr = ((wy << 3) + (t >> 3) + 4) & 255;
                int gc = ((wx << 3) + (t & 7) + 4) & 255;
                ob[((size_t)((gr << 8) + gc) << 7) + n] = (*accs[mt])[r] + bias;
            }
    }
}

extern "C" void kernel_launch(void* const* d_in, const int* in_sizes, int n_in,
                              void* d_out, int out_size, void* d_ws, size_t ws_size,
                              hipStream_t stream) {
    (void)in_sizes; (void)n_in; (void)out_size; (void)ws_size;
    const float* x     = (const float*)d_in[0];
    const float* w_qkv = (const float*)d_in[1];
    const float* pos   = (const float*)d_in[2];
    const float* w_out = (const float*)d_in[3];
    const float* b_out = (const float*)d_in[4];
    float* out = (float*)d_out;

    _Float16* wqkv_h = (_Float16*)d_ws;                 // 49152 halfs = 96 KB
    _Float16* wout_h = wqkv_h + WQKV_ELEMS;             // 16384 halfs = 32 KB

    hipLaunchKernelGGL(weight_swizzle_kernel, dim3(256), dim3(256), 0, stream,
                       w_qkv, w_out, wqkv_h, wout_h);

    dim3 grid(8 * 1024);   // 8 batches * 32*32 windows
    dim3 block(256);       // 8 wave32 waves
    hipLaunchKernelGGL(swin_window_attn_kernel, grid, block, SMEM_BYTES, stream,
                       x, wqkv_h, pos, wout_h, b_out, out);
}
